// TestCUDAMoe_54305566491400
// MI455X (gfx1250) — matrix operand order repaired and checked by reading, not compiled, not run on hardware
//
#include <hip/hip_runtime.h>
#include <cstdint>

#define H_DIM 4096
#define NEXP  16
#define INTER 448
#define EI    (NEXP * INTER)

typedef __attribute__((ext_vector_type(16))) __bf16          v16bf;
typedef __attribute__((ext_vector_type(8)))  float           v8f;
typedef __attribute__((ext_vector_type(8)))  unsigned short  u16x8;

union Frag16 {
    v16bf          v;
    __bf16         b[16];
    unsigned short s[16];
    u16x8          h2[2];
};

__device__ inline v8f zero8() {
    v8f z = {0.f, 0.f, 0.f, 0.f, 0.f, 0.f, 0.f, 0.f};
    return z;
}

__device__ inline v8f wmma_bf16(v16bf a, v16bf b, v8f c) {
    return __builtin_amdgcn_wmma_f32_16x16x32_bf16(
        /*neg_a=*/false, a, /*neg_b=*/false, b,
        /*c_mod=*/(short)0, c, /*reuse_a=*/false, /*reuse_b=*/false);
}

// Load a 16-element bf16 fragment: two 8-element chunks at element offsets
// row+k0 / row+k1. PRE=true -> bf16 in memory (two b128 loads);
// PRE=false -> f32 in memory + hardware f32->bf16 convert.
template <bool PRE>
__device__ inline v16bf loadfrag(const void* __restrict__ base, unsigned row,
                                 int k0, int k1) {
    Frag16 f;
    if constexpr (PRE) {
        const unsigned short* p = (const unsigned short*)base + row;
        f.h2[0] = *(const u16x8*)(p + k0);
        f.h2[1] = *(const u16x8*)(p + k1);
    } else {
        const float* p = (const float*)base + row;
#pragma unroll
        for (int j = 0; j < 8; ++j) f.b[j] = (__bf16)p[k0 + j];
#pragma unroll
        for (int j = 0; j < 8; ++j) f.b[8 + j] = (__bf16)p[k1 + j];
    }
    return f.v;
}

// ---------------------------------------------------------------------------
// Kernel 0: bulk f32 -> bf16 conversion (8 elements/thread, b128 in/out)
// ---------------------------------------------------------------------------
__global__ void convert_bf16_kernel(const float* __restrict__ src,
                                    unsigned short* __restrict__ dst,
                                    long long n) {
    const long long i =
        (((long long)blockIdx.x * blockDim.x) + threadIdx.x) * 8;
    if (i + 8 > n) return;
    const float4 f0 = *(const float4*)(src + i);
    const float4 f1 = *(const float4*)(src + i + 4);
    union { u16x8 v; __bf16 b[8]; } o;
    o.b[0] = (__bf16)f0.x; o.b[1] = (__bf16)f0.y;
    o.b[2] = (__bf16)f0.z; o.b[3] = (__bf16)f0.w;
    o.b[4] = (__bf16)f1.x; o.b[5] = (__bf16)f1.y;
    o.b[6] = (__bf16)f1.z; o.b[7] = (__bf16)f1.w;
    *(u16x8*)(dst + i) = o.v;
}

// ---------------------------------------------------------------------------
// Kernel 1: router. One wave per token; 16 logits -> top-2 -> 2-way softmax.
// ---------------------------------------------------------------------------
__global__ void moe_router_kernel(const float* __restrict__ x,
                                  const float* __restrict__ gw,
                                  int* __restrict__ topk_idx,
                                  float* __restrict__ topk_w, int bs) {
    const int wv   = blockIdx.x * (blockDim.x >> 5) + (threadIdx.x >> 5);
    const int lane = threadIdx.x & 31;
    if (wv >= bs) return;
    const float* xr = x + (size_t)wv * H_DIM;

    float acc[NEXP];
#pragma unroll
    for (int e = 0; e < NEXP; ++e) acc[e] = 0.f;

    for (int h = lane; h < H_DIM; h += 32) {
        const float xv = xr[h];
#pragma unroll
        for (int e = 0; e < NEXP; ++e)
            acc[e] = fmaf(xv, gw[e * H_DIM + h], acc[e]);
    }
#pragma unroll
    for (int e = 0; e < NEXP; ++e) {
#pragma unroll
        for (int off = 16; off >= 1; off >>= 1)
            acc[e] += __shfl_xor(acc[e], off, 32);
    }

    int i1 = 0; float v1 = acc[0];
#pragma unroll
    for (int e = 1; e < NEXP; ++e)
        if (acc[e] > v1) { v1 = acc[e]; i1 = e; }
    int i2 = -1; float v2 = -3.4e38f;
#pragma unroll
    for (int e = 0; e < NEXP; ++e)
        if (e != i1 && acc[e] > v2) { v2 = acc[e]; i2 = e; }

    const float w1 = 1.f / (1.f + __expf(v2 - v1));
    if (lane == 0) {
        topk_idx[2 * wv]     = i1;
        topk_idx[2 * wv + 1] = i2;
        topk_w[2 * wv]       = w1;
        topk_w[2 * wv + 1]   = 1.f - w1;
    }
}

// ---------------------------------------------------------------------------
// Kernel 2: deterministic per-expert compaction (ballot prefix-scan).
// ---------------------------------------------------------------------------
__global__ void moe_compact_kernel(const int* __restrict__ topk_idx,
                                   const float* __restrict__ topk_w,
                                   int* __restrict__ counts,
                                   int* __restrict__ lists,
                                   float* __restrict__ wlists, int bs) {
    const int e    = blockIdx.x;
    const int lane = threadIdx.x;
    int base = 0;
    for (int start = 0; start < bs; start += 32) {
        const int t = start + lane;
        bool  ok = false;
        float wv = 0.f;
        if (t < bs) {
            const int i0 = topk_idx[2 * t];
            const int i1 = topk_idx[2 * t + 1];
            if (i0 == e)      { ok = true; wv = topk_w[2 * t]; }
            else if (i1 == e) { ok = true; wv = topk_w[2 * t + 1]; }
        }
        const unsigned mask = __builtin_amdgcn_ballot_w32(ok);
        const int pre = __popc(mask & ((1u << lane) - 1u));
        if (ok) {
            lists[e * bs + base + pre]  = t;
            wlists[e * bs + base + pre] = wv;
        }
        base += __popc(mask);
    }
    if (lane == 0) counts[e] = base;
}

// ---------------------------------------------------------------------------
// Kernel 3: expert FFN. One block = one (expert, 16-token tile), 224 threads
// (7 waves x 4 N-tiles = the 28 tiles of INTER=448).
//   Phase 1: h = w * silu(x@g_e^T) * (x@u_e^T)   -> LDS (bf16)
//   Phase 2: out += h @ d_e^T                     (atomic f32 accumulate)
// Per-wave working set kept small (8 accumulators + 9 batched fragments) so
// the scheduler can keep ~18 b128 loads in flight across each WMMA burst.
// ---------------------------------------------------------------------------
template <bool PRE>
__global__ __launch_bounds__(224) void moe_ffn_kernel(
    const void* __restrict__ x, const void* __restrict__ uw,
    const void* __restrict__ gmat, const void* __restrict__ dw,
    const int* __restrict__ counts, const int* __restrict__ lists,
    const float* __restrict__ wlists, float* __restrict__ out, int bs) {
    const int e  = blockIdx.y;
    const int mt = blockIdx.x;
    const int cnt = counts[e];
    if (mt * 16 >= cnt) return;
    const int valid = min(16, cnt - mt * 16);

    __shared__ int   tok_s[16];
    __shared__ float w_s[16];
    // Row padded 448->456 halves (912B = 57*16): 16B-aligned, bank-spread
    // b128 fragment reads.
    __shared__ __align__(16) __bf16 h_lds[16][INTER + 8];

    if (threadIdx.x < 16) {
        const int r   = threadIdx.x;
        const int src = e * bs + mt * 16 + ((r < valid) ? r : 0);
        tok_s[r] = lists[src];
        w_s[r]   = (r < valid) ? wlists[src] : 0.f;
    }
    __syncthreads();

    const int wv   = threadIdx.x >> 5;   // wave 0..6
    const int lane = threadIdx.x & 31;
    const int hi   = lane >> 4;          // lane half
    const int lid  = lane & 15;

    // 16-bit A fragment K-chunks per lane half; B fragment K offset.
    const int a0   = hi ? 16 : 0;
    const int a1   = hi ? 24 : 8;
    const int bofs = hi * 16;

    // ---------------- phase 1: up/gate GEMMs over K=4096 ----------------
    const unsigned xoff = (unsigned)tok_s[lid] * H_DIM;

    unsigned grow[4];
#pragma unroll
    for (int t = 0; t < 4; ++t)
        grow[t] = (unsigned)(e * INTER + (wv * 4 + t) * 16 + lid) * H_DIM;

    v8f accg[4], accu[4];
#pragma unroll
    for (int t = 0; t < 4; ++t) { accg[t] = zero8(); accu[t] = zero8(); }

#pragma unroll 1
    for (int ks = 0; ks < H_DIM / 32; ++ks) {
        const int kb = ks * 32;
        // batch-issue all 18 b128 loads of this K-step...
        const v16bf a = loadfrag<PRE>(x, xoff, kb + a0, kb + a1);
        v16bf bg[4], bu[4];
#pragma unroll
        for (int t = 0; t < 4; ++t) {
            bg[t] = loadfrag<PRE>(gmat, grow[t], kb + bofs, kb + bofs + 8);
            bu[t] = loadfrag<PRE>(uw,  grow[t], kb + bofs, kb + bofs + 8);
        }
        // ...then burst 8 WMMAs (load-issue order -> partial loadcnt waits)
#pragma unroll
        for (int t = 0; t < 4; ++t) {
            accg[t] = wmma_bf16(a, bg[t], accg[t]);
            accu[t] = wmma_bf16(a, bu[t], accu[t]);
        }
    }

    // silu(gate)*up, fold routing weight, store bf16 tile to LDS
#pragma unroll
    for (int t = 0; t < 4; ++t) {
        const int nc = (wv * 4 + t) * 16 + lid;
#pragma unroll
        for (int r = 0; r < 8; ++r) {
            const int   m  = r + hi * 8;
            const float gv = accg[t][r];
            const float hv = (gv / (1.f + __expf(-gv))) * accu[t][r] * w_s[m];
            h_lds[m][nc] = (__bf16)hv;
        }
    }
    __syncthreads();

    // ---------------- phase 2: down projection, N = 4096 ----------------
    // 64 groups of 4 N-tiles; group c handled by wave (c mod 7). Per K-step:
    // 1 LDS A fragment + 4 global B fragments -> 4 WMMAs.
#pragma unroll 1
    for (int c = wv; c < 64; c += 7) {
        const int nt0 = c * 4;
        unsigned ncol[4], drow[4];
        v8f acc[4];
#pragma unroll
        for (int i = 0; i < 4; ++i) {
            ncol[i] = (unsigned)((nt0 + i) * 16 + lid);
            drow[i] = ncol[i] * (unsigned)EI + (unsigned)(e * INTER);
            acc[i]  = zero8();
        }
#pragma unroll 1
        for (int ks = 0; ks < INTER / 32; ++ks) {
            const int kb = ks * 32;
            Frag16 af;
            af.h2[0] = *(const u16x8*)&h_lds[lid][kb + a0];
            af.h2[1] = *(const u16x8*)&h_lds[lid][kb + a1];
            v16bf b[4];
#pragma unroll
            for (int i = 0; i < 4; ++i)
                b[i] = loadfrag<PRE>(dw, drow[i], kb + bofs, kb + bofs + 8);
#pragma unroll
            for (int i = 0; i < 4; ++i)
                acc[i] = wmma_bf16(af.v, b[i], acc[i]);
        }
#pragma unroll
        for (int i = 0; i < 4; ++i) {
#pragma unroll
            for (int r = 0; r < 8; ++r) {
                const int m = r + hi * 8;
                if (m < valid)
                    atomicAdd(&out[(size_t)tok_s[m] * H_DIM + ncol[i]],
                              acc[i][r]);
            }
        }
    }
}

// ---------------------------------------------------------------------------
extern "C" void kernel_launch(void* const* d_in, const int* in_sizes, int n_in,
                              void* d_out, int out_size, void* d_ws, size_t ws_size,
                              hipStream_t stream) {
    const float* x    = (const float*)d_in[0];  // hid    [bs, 4096]
    const float* gw   = (const float*)d_in[1];  // gate_w [16, 4096]
    const float* uw   = (const float*)d_in[2];  // u      [7168, 4096]
    const float* gmat = (const float*)d_in[3];  // g      [7168, 4096]
    const float* dw   = (const float*)d_in[4];  // d      [4096, 7168]
    float* out = (float*)d_out;

    const int bs = in_sizes[0] / H_DIM;

    // ---- workspace carve-out ----
    char*  base = (char*)d_ws;
    size_t off  = 0;
    auto carve = [&](size_t bytes) {
        char* p = base + off;
        off = (off + bytes + 15) & ~(size_t)15;
        return p;
    };
    int*   topk_idx = (int*)carve((size_t)bs * 2 * sizeof(int));
    float* topk_w   = (float*)carve((size_t)bs * 2 * sizeof(float));
    int*   counts   = (int*)carve(64);
    int*   lists    = (int*)carve((size_t)NEXP * bs * sizeof(int));
    float* wlists   = (float*)carve((size_t)NEXP * bs * sizeof(float));

    const long long nx = (long long)bs * H_DIM;
    const long long nw = (long long)EI * H_DIM;  // u, g, and d element counts
    unsigned short* xb = (unsigned short*)carve((size_t)nx * 2);
    unsigned short* ub = (unsigned short*)carve((size_t)nw * 2);
    unsigned short* gb = (unsigned short*)carve((size_t)nw * 2);
    unsigned short* db = (unsigned short*)carve((size_t)nw * 2);
    const bool pre = (off <= ws_size);  // constant per run -> deterministic

    hipMemsetAsync(d_out, 0, (size_t)out_size * sizeof(float), stream);

    moe_router_kernel<<<dim3((bs + 3) / 4), dim3(128), 0, stream>>>(
        x, gw, topk_idx, topk_w, bs);
    moe_compact_kernel<<<dim3(NEXP), dim3(32), 0, stream>>>(
        topk_idx, topk_w, counts, lists, wlists, bs);

    const int mtiles = (bs + 15) / 16;
    if (pre) {
        const int cvt_blk = 256;
        auto cvt_grid = [&](long long n) {
            return dim3((unsigned)((n / 8 + cvt_blk - 1) / cvt_blk));
        };
        convert_bf16_kernel<<<cvt_grid(nx), cvt_blk, 0, stream>>>(x, xb, nx);
        convert_bf16_kernel<<<cvt_grid(nw), cvt_blk, 0, stream>>>(uw, ub, nw);
        convert_bf16_kernel<<<cvt_grid(nw), cvt_blk, 0, stream>>>(gmat, gb, nw);
        convert_bf16_kernel<<<cvt_grid(nw), cvt_blk, 0, stream>>>(dw, db, nw);
        moe_ffn_kernel<true><<<dim3(mtiles, NEXP), dim3(224), 0, stream>>>(
            xb, ub, gb, db, counts, lists, wlists, out, bs);
    } else {
        moe_ffn_kernel<false><<<dim3(mtiles, NEXP), dim3(224), 0, stream>>>(
            x, uw, gmat, dw, counts, lists, wlists, out, bs);
    }
}
